// MoEV2_29703993819797
// MI455X (gfx1250) — matrix operand order repaired and checked by reading, not compiled, not run on hardware
//
#include <hip/hip_runtime.h>

// ---------------------------------------------------------------------------
// MoE layer for MI455X (gfx1250, wave32, WMMA + async-to-LDS, double-buffered).
// Sparse top-2 routing (137 GFLOP vs 549 dense). bf16 WMMA 16x16x32 core.
// Weights pre-converted fp32->bf16 AND pre-transposed to [n][k] once, so GEMM
// tile staging is contiguous bf16 -> LDS via GLOBAL_LOAD_ASYNC_TO_LDS_B128.
// LDS is double-buffered: tile k+1 streams in (ASYNCcnt<=4) while tile k runs
// through v_wmma, so the hot loop is async-issue + ds_load_b128 + wmma only.
// ---------------------------------------------------------------------------

typedef __attribute__((ext_vector_type(16))) __bf16 v16bf;
typedef __attribute__((ext_vector_type(8)))  __bf16 v8bf;
typedef __attribute__((ext_vector_type(8)))  float  v8f;

constexpr int   D        = 1024;
constexpr int   H        = 4096;
constexpr int   E        = 8;
constexpr float EPS      = 1e-5f;
constexpr float INV_TEMP = 1.0f / 1.5f;

union BF16x16 { v16bf v; v8bf h[2]; };
union F32x8   { v8f   v; float f[8]; };

__device__ __forceinline__ uint32_t lds_off(const void* p) {
  // Generic LDS addresses carry the wave-relative LDS byte offset in [31:0].
  return (uint32_t)(uintptr_t)p;
}

__device__ __forceinline__ void async_copy_b128(uint32_t dst_lds, const void* gsrc) {
  asm volatile("global_load_async_to_lds_b128 %0, %1, off"
               :: "v"(dst_lds), "v"(gsrc) : "memory");
}

__device__ __forceinline__ void wait_async0() {
  asm volatile("s_wait_asynccnt 0x0" ::: "memory");
}
__device__ __forceinline__ void wait_async4() {   // tile k done, k+1 in flight
  asm volatile("s_wait_asynccnt 0x4" ::: "memory");
}

// ---------------------------------------------------------------------------
// Kernel 0: fp32 -> bf16 convert + transpose weights:  in[e][r][c] -> out[e][c][r]
// ---------------------------------------------------------------------------
__global__ __launch_bounds__(256)
void transpose_cvt_kernel(const float* __restrict__ in, __bf16* __restrict__ outp,
                          int R, int C)
{
  __shared__ float tile[32][33];
  const int e  = blockIdx.z;
  const int c0 = blockIdx.x * 32;
  const int r0 = blockIdx.y * 32;
  const int tx = threadIdx.x & 31;
  const int ty = threadIdx.x >> 5;  // 0..7
  const float* src = in   + (size_t)e * R * C;
  __bf16*      dst = outp + (size_t)e * R * C;
  #pragma unroll
  for (int i = 0; i < 4; ++i)
    tile[ty + 8 * i][tx] = src[(size_t)(r0 + ty + 8 * i) * C + c0 + tx];
  __syncthreads();
  #pragma unroll
  for (int i = 0; i < 4; ++i)
    dst[(size_t)(c0 + ty + 8 * i) * R + r0 + tx] = (__bf16)tile[tx][ty + 8 * i];
}

// ---------------------------------------------------------------------------
// Kernel 1: LayerNorm + router softmax/top-2. One block (256 thr) per token.
// ---------------------------------------------------------------------------
__global__ __launch_bounds__(256)
void ln_router_kernel(const float* __restrict__ x,
                      const float* __restrict__ gamma,
                      const float* __restrict__ beta,
                      const float* __restrict__ rw,
                      float* __restrict__ out,     // residual-initialized output
                      __bf16* __restrict__ xn_bf,  // normalized activations, bf16
                      float* __restrict__ wdense)  // [NT][E] routing weights
{
  const int t    = blockIdx.x;
  const int tid  = threadIdx.x;
  const int lane = tid & 31;
  const int wid  = tid >> 5;

  __shared__ float xn_s[D];
  __shared__ float red_s[16];
  __shared__ float logits_s[E];

  const float4 xv = ((const float4*)(x + (size_t)t * D))[tid];
  float s  = xv.x + xv.y + xv.z + xv.w;
  float ss = xv.x * xv.x + xv.y * xv.y + xv.z * xv.z + xv.w * xv.w;
  #pragma unroll
  for (int off = 16; off > 0; off >>= 1) {
    s  += __shfl_down(s, off, 32);
    ss += __shfl_down(ss, off, 32);
  }
  if (lane == 0) { red_s[wid] = s; red_s[8 + wid] = ss; }
  __syncthreads();

  float mu = 0.f, m2 = 0.f;
  #pragma unroll
  for (int i = 0; i < 8; ++i) { mu += red_s[i]; m2 += red_s[8 + i]; }
  mu *= (1.0f / (float)D);
  const float rstd = rsqrtf(m2 * (1.0f / (float)D) - mu * mu + EPS);

  const float4 gv = ((const float4*)gamma)[tid];
  const float4 bv = ((const float4*)beta)[tid];
  float4 xn;
  xn.x = (xv.x - mu) * rstd * gv.x + bv.x;
  xn.y = (xv.y - mu) * rstd * gv.y + bv.y;
  xn.z = (xv.z - mu) * rstd * gv.z + bv.z;
  xn.w = (xv.w - mu) * rstd * gv.w + bv.w;

  const int d0 = tid * 4;
  xn_s[d0 + 0] = xn.x; xn_s[d0 + 1] = xn.y;
  xn_s[d0 + 2] = xn.z; xn_s[d0 + 3] = xn.w;

  union { __bf16 b[4]; uint2 u; } pk;
  pk.b[0] = (__bf16)xn.x; pk.b[1] = (__bf16)xn.y;
  pk.b[2] = (__bf16)xn.z; pk.b[3] = (__bf16)xn.w;
  ((uint2*)(xn_bf + (size_t)t * D))[tid] = pk.u;

  ((float4*)(out + (size_t)t * D))[tid] = xv;  // residual
  __syncthreads();

  // Router: wave `wid` computes logit for expert `wid` (8 waves == 8 experts).
  float acc = 0.f;
  const float* rwe = rw + (size_t)wid * D;
  for (int i = lane; i < D; i += 32) acc += xn_s[i] * rwe[i];
  #pragma unroll
  for (int off = 16; off > 0; off >>= 1) acc += __shfl_down(acc, off, 32);
  if (lane == 0) logits_s[wid] = acc;
  __syncthreads();

  if (tid == 0) {
    float l[E], p[E];
    float mx = -1e30f;
    #pragma unroll
    for (int e = 0; e < E; ++e) { l[e] = logits_s[e] * INV_TEMP; mx = fmaxf(mx, l[e]); }
    float sum = 0.f;
    #pragma unroll
    for (int e = 0; e < E; ++e) { p[e] = __expf(l[e] - mx); sum += p[e]; }
    const float inv = 1.f / sum;
    #pragma unroll
    for (int e = 0; e < E; ++e) p[e] *= inv;
    int i1 = 0;
    #pragma unroll
    for (int e = 1; e < E; ++e) if (p[e] > p[i1]) i1 = e;   // lowest index on tie
    int i2 = (i1 == 0) ? 1 : 0;
    #pragma unroll
    for (int e = 0; e < E; ++e) if (e != i1 && p[e] > p[i2]) i2 = e;
    #pragma unroll
    for (int e = 0; e < E; ++e)
      wdense[(size_t)t * E + e] = (e == i1) ? p[i1] : ((e == i2) ? p[i2] : 0.f);
  }
}

// ---------------------------------------------------------------------------
// Kernel 2: deterministic per-expert compaction (block prefix scan, no atomics)
// ---------------------------------------------------------------------------
__global__ __launch_bounds__(256)
void build_lists_kernel(const float* __restrict__ wdense,
                        int* __restrict__ counts,
                        int* __restrict__ tok_list,   // [E][NT]
                        float* __restrict__ w_list,   // [E][NT]
                        int NT)
{
  const int e   = blockIdx.x;
  const int tid = threadIdx.x;
  __shared__ int ps[256];
  int base = 0;
  for (int t0 = 0; t0 < NT; t0 += 256) {
    const int t = t0 + tid;
    const float w = (t < NT) ? wdense[(size_t)t * E + e] : 0.f;
    const int flag = (w > 0.f) ? 1 : 0;
    ps[tid] = flag;
    __syncthreads();
    for (int off = 1; off < 256; off <<= 1) {   // Hillis-Steele inclusive scan
      int v = 0;
      if (tid >= off) v = ps[tid - off];
      __syncthreads();
      ps[tid] += v;
      __syncthreads();
    }
    if (flag) {
      const int slot = base + ps[tid] - 1;
      tok_list[(size_t)e * NT + slot] = t;
      w_list[(size_t)e * NT + slot]   = w;
    }
    base += ps[255];
    __syncthreads();
  }
  if (tid == 0) counts[e] = base;
}

__global__ void scan_offsets_kernel(const int* __restrict__ counts,
                                    int* __restrict__ offsets)
{
  if (threadIdx.x == 0 && blockIdx.x == 0) {
    int acc = 0;
    for (int e = 0; e < E; ++e) { offsets[e] = acc; acc += counts[e]; }
  }
}

// ---------------------------------------------------------------------------
// Fused grouped-GEMM core: 128x128 tile, K-steps of 32, bf16 WMMA 16x16x32.
// 8 waves in a 2(M) x 4(N) grid, each owning a 64x32 subtile (4x2 fragments).
// Double-buffered LDS; tiles staged with GLOBAL_LOAD_ASYNC_TO_LDS_B128 and
// overlapped with WMMA via s_wait_asynccnt 0x4 (4 copies/thread/tile, issued
// unconditionally: padded rows are clamped and their outputs discarded).
// STAGE 1: h = silu(xn @ W1[e])        (A gathered by token list)
// STAGE 2: out += w * (h @ W2[e])      (A contiguous in packed h buffer)
// ---------------------------------------------------------------------------
template<int STAGE>
__global__ __launch_bounds__(256)
void moe_gemm_kernel(const __bf16* __restrict__ Bt,    // W1t/W2t bf16 [e][n][k]
                     const __bf16* __restrict__ Abase, // xn_bf or hbuf
                     __bf16* __restrict__ hbuf,
                     float* __restrict__ outp,
                     const int* __restrict__ counts,
                     const int* __restrict__ offsets,
                     const int* __restrict__ tok_list,
                     const float* __restrict__ w_list,
                     int NT)
{
  constexpr int TM = 128, TN = 128, TK = 32;
  constexpr int KD     = (STAGE == 1) ? D : H;  // reduction dim (= Bt row stride)
  constexpr int NDIM   = (STAGE == 1) ? H : D;  // output cols
  constexpr int NTIL   = NDIM / TN;
  constexpr int KSTEPS = KD / TK;
  constexpr uint32_t BUFB = TM * TK * 2;        // bytes per LDS tile buffer

  const int mtiles = NT / TM;
  const int bx = blockIdx.x;
  const int e  = bx / (mtiles * NTIL);
  const int rm = bx % (mtiles * NTIL);
  const int m0 = (rm / NTIL) * TM;
  const int n0 = (rm % NTIL) * TN;

  const int cnt = counts[e];
  if (m0 >= cnt) return;                        // dynamic sparsity: skip empty tiles
  const int aoff = offsets[e];

  const int tid  = threadIdx.x;
  const int lane = tid & 31;
  const int wid  = tid >> 5;
  const int wm   = (wid & 1) * 64;              // wave M base in tile
  const int wn   = (wid >> 1) * 32;             // wave N base in tile
  const int fr   = lane & 15;                   // row/col within fragment
  const int kh   = lane >> 4;                   // lane half (K split per ISA 7.12.2)

  __shared__ __bf16 Asub[2][TM][TK];            // [buf][m][k]
  __shared__ __bf16 Bsub[2][TN][TK];            // [buf][n][k]
  __shared__ int    tok_s[TM];
  __shared__ float  w_s[TM];

  if (STAGE == 2) {                             // epilogue routing info
    for (int r = tid; r < TM; r += 256) {
      const int slot = m0 + r;
      tok_s[r] = (slot < cnt) ? tok_list[(size_t)e * NT + slot] : 0;
      w_s[r]   = (slot < cnt) ? w_list[(size_t)e * NT + slot]   : 0.f;
    }
  }

  const __bf16* Bge = Bt + (size_t)e * D * H;   // expert's [NDIM][KD] bf16 matrix

  // --- per-thread staging chunks: 2 A + 2 B 16B copies per tile, fixed rows ---
  const int c0i = tid * 2,      c1i = tid * 2 + 1;
  const int row0 = c0i >> 2,    seg0 = c0i & 3;
  const int row1 = c1i >> 2,    seg1 = c1i & 3;
  const uint32_t off0 = (uint32_t)(row0 * 64 + seg0 * 16);
  const uint32_t off1 = (uint32_t)(row1 * 64 + seg1 * 16);

  // Clamp padded rows to a valid row; their outputs are masked in the epilogue.
  const int slot0 = (m0 + row0 < cnt) ? (m0 + row0) : (cnt - 1);
  const int slot1 = (m0 + row1 < cnt) ? (m0 + row1) : (cnt - 1);
  const __bf16* pa0;
  const __bf16* pa1;
  if (STAGE == 1) {
    pa0 = Abase + (size_t)tok_list[(size_t)e * NT + slot0] * D + seg0 * 8;
    pa1 = Abase + (size_t)tok_list[(size_t)e * NT + slot1] * D + seg1 * 8;
  } else {
    pa0 = Abase + (size_t)(aoff + slot0) * H + seg0 * 8;
    pa1 = Abase + (size_t)(aoff + slot1) * H + seg1 * 8;
  }
  const __bf16* pb0 = Bge + (size_t)(n0 + row0) * KD + seg0 * 8;
  const __bf16* pb1 = Bge + (size_t)(n0 + row1) * KD + seg1 * 8;

  const uint32_t asub0 = lds_off(&Asub[0][0][0]);
  const uint32_t bsub0 = lds_off(&Bsub[0][0][0]);

  auto issue_tile = [&](int buf) {
    const uint32_t ab = asub0 + (uint32_t)buf * BUFB;
    const uint32_t bb = bsub0 + (uint32_t)buf * BUFB;
    async_copy_b128(ab + off0, pa0);
    async_copy_b128(ab + off1, pa1);
    async_copy_b128(bb + off0, pb0);
    async_copy_b128(bb + off1, pb1);
    pa0 += TK; pa1 += TK; pb0 += TK; pb1 += TK;   // advance 64 B along K
  };

  F32x8 acc[4][2];
  #pragma unroll
  for (int i = 0; i < 4; ++i)
    #pragma unroll
    for (int j = 0; j < 2; ++j)
      #pragma unroll
      for (int r = 0; r < 8; ++r) acc[i][j].f[r] = 0.f;

  issue_tile(0);                                 // prologue
  for (int step = 0; step < KSTEPS; ++step) {
    const int cur = step & 1;
    const bool more = (step + 1 < KSTEPS);
    if (more) issue_tile(cur ^ 1);               // stream next tile behind compute
    if (more) wait_async4(); else wait_async0(); // current tile's 4 copies done
    __syncthreads();

    // ---- fragments (ISA 7.12.2 16-bit layouts) ----
    BF16x16 a[4];
    #pragma unroll
    for (int i = 0; i < 4; ++i) {
      const int r = wm + i * 16 + fr;            // A lane: K {0..7,16..23} | {8..15,24..31}
      a[i].h[0] = *(const v8bf*)&Asub[cur][r][kh * 8];
      a[i].h[1] = *(const v8bf*)&Asub[cur][r][16 + kh * 8];
    }
    BF16x16 b[2];
    #pragma unroll
    for (int j = 0; j < 2; ++j) {
      const int ccol = wn + j * 16 + fr;         // B lane: K {0..7,16..23} | {8..15,24..31}
      b[j].h[0] = *(const v8bf*)&Bsub[cur][ccol][kh * 8];
      b[j].h[1] = *(const v8bf*)&Bsub[cur][ccol][16 + kh * 8];
    }
    #pragma unroll
    for (int i = 0; i < 4; ++i)
      #pragma unroll
      for (int j = 0; j < 2; ++j)
        acc[i][j].v = __builtin_amdgcn_wmma_f32_16x16x32_bf16(
            false, a[i].v, false, b[j].v, (short)0, acc[i][j].v, false, false);

    __syncthreads();                             // buffer free before re-fill
  }

  // ---- epilogue (C/D layout: lane holds col fr, rows kh*8 .. kh*8+7) ----
  #pragma unroll
  for (int i = 0; i < 4; ++i) {
    const int mb = wm + i * 16 + kh * 8;
    #pragma unroll
    for (int j = 0; j < 2; ++j) {
      const int n = n0 + wn + j * 16 + fr;
      #pragma unroll
      for (int r = 0; r < 8; ++r) {
        const int mrow = mb + r;
        const int slot = m0 + mrow;
        if (slot < cnt) {
          float v = acc[i][j].f[r];
          if (STAGE == 1) {
            v = v / (1.f + __expf(-v));          // SiLU
            hbuf[(size_t)(aoff + slot) * H + n] = (__bf16)v;
          } else {
            atomicAdd(outp + (size_t)tok_s[mrow] * D + n, w_s[mrow] * v);
          }
        }
      }
    }
  }
}

// ---------------------------------------------------------------------------
extern "C" void kernel_launch(void* const* d_in, const int* in_sizes, int n_in,
                              void* d_out, int out_size, void* d_ws, size_t ws_size,
                              hipStream_t stream)
{
  const float* x     = (const float*)d_in[0];
  const float* gamma = (const float*)d_in[1];
  const float* beta  = (const float*)d_in[2];
  const float* rw    = (const float*)d_in[3];
  const float* W1    = (const float*)d_in[4];
  const float* W2    = (const float*)d_in[5];
  float* out = (float*)d_out;
  const int NT = in_sizes[0] / D;   // 4096 tokens

  // Workspace layout (~201 MB total):
  char*   ws      = (char*)d_ws;
  int*    counts  = (int*)ws;                                   // E
  int*    offsets = (int*)(ws + 128);                           // E
  float*  wdense  = (float*)(ws + 256);                         // NT*E fp32
  int*    tokl    = (int*)(ws + 256 + (size_t)4  * NT * E);     // NT*E int
  float*  wl      = (float*)(ws + 256 + (size_t)8  * NT * E);   // NT*E fp32
  __bf16* xn_bf   = (__bf16*)(ws + 256 + (size_t)12 * NT * E);  // NT*D bf16
  __bf16* hbuf    = xn_bf + (size_t)NT * D;                     // 2*NT*H bf16
  __bf16* W1t     = hbuf + (size_t)2 * NT * H;                  // E*H*D bf16 [e][f][d]
  __bf16* W2t     = W1t + (size_t)E * D * H;                    // E*D*H bf16 [e][d][f]
  (void)n_in; (void)out_size; (void)ws_size;

  // Pre-convert + pre-transpose weights to bf16 [n][k] (once per launch).
  transpose_cvt_kernel<<<dim3(H / 32, D / 32, E), 256, 0, stream>>>(W1, W1t, D, H);
  transpose_cvt_kernel<<<dim3(D / 32, H / 32, E), 256, 0, stream>>>(W2, W2t, H, D);

  ln_router_kernel<<<NT, 256, 0, stream>>>(x, gamma, beta, rw, out, xn_bf, wdense);
  build_lists_kernel<<<E, 256, 0, stream>>>(wdense, counts, tokl, wl, NT);
  scan_offsets_kernel<<<1, 1, 0, stream>>>(counts, offsets);

  const int mt = NT / 128;
  moe_gemm_kernel<1><<<E * mt * (H / 128), 256, 0, stream>>>(
      W1t, xn_bf, hbuf, out, counts, offsets, tokl, wl, NT);
  moe_gemm_kernel<2><<<E * mt * (D / 128), 256, 0, stream>>>(
      W2t, hbuf, hbuf, out, counts, offsets, tokl, wl, NT);
}
